// MQALayer_2791728742932
// MI455X (gfx1250) — matrix operand order repaired and checked by reading, not compiled, not run on hardware
//
#include <hip/hip_runtime.h>
#include <hip/hip_bf16.h>
#include <stdint.h>

// ---------------- problem constants ----------------
#define T_TOK 8192
#define DIM   7168
#define NOUT  512            // 2*2*HEAD_DIM
#define S_OUT 2048           // T / RATIO

// ---------------- GEMM tiling ----------------
#define BM 64
#define BN 256
#define BK 32
#define NKB (DIM / BK)       // 224
#define LDSS 40              // 32 bf16 + 8 pad -> 80B rows (16B aligned, conflict-free b128)

typedef __attribute__((ext_vector_type(16))) __bf16 v16bf;
typedef __attribute__((ext_vector_type(8)))  float  v8f;

union FragU { v16bf v; uint4 q[2]; };

// full RNE hi/lo split (used only in the bandwidth-bound pre-convert pass)
__device__ __forceinline__ void bf16_split(float x, unsigned& hi, unsigned& lo) {
    unsigned u  = __float_as_uint(x);
    unsigned rh = u + 0x7FFFu + ((u >> 16) & 1u);
    hi = rh >> 16;
    float fh = __uint_as_float(rh & 0xFFFF0000u);
    float d  = x - fh;
    unsigned v  = __float_as_uint(d);
    unsigned rl = v + 0x7FFFu + ((v >> 16) & 1u);
    lo = rl >> 16;
}

// fast in-loop split: hi = truncate(x) (residual captured exactly in lo, RNE);
// pairs packed with v_perm_b32 to avoid 16-bit half-register shuffles.
__device__ __forceinline__ void split_pack2(float a, float b, unsigned& hiPair, unsigned& loPair) {
    const unsigned ua = __float_as_uint(a);
    const unsigned ub = __float_as_uint(b);
    hiPair = __builtin_amdgcn_perm(ub, ua, 0x07060302u);      // {b.hi16, a.hi16}
    const float raf = a - __uint_as_float(ua & 0xFFFF0000u);
    const float rbf = b - __uint_as_float(ub & 0xFFFF0000u);
    unsigned va = __float_as_uint(raf);
    unsigned vb = __float_as_uint(rbf);
    va = va + 0x7FFFu + ((va >> 16) & 1u);
    vb = vb + 0x7FFFu + ((vb >> 16) & 1u);
    loPair = __builtin_amdgcn_perm(vb, va, 0x07060302u);
}

// CDNA5 async copy, GVS form: SGPR base + 32-bit VGPR offset + imm offset
// (imm offset applies to BOTH the LDS dest and the memory source, ISA §4.4)
#define ASYNC_B128(ldsv, voffv, sbase, OFFSTR)                               \
    asm volatile("global_load_async_to_lds_b128 %0, %1, %2 offset:" OFFSTR   \
                 :: "v"(ldsv), "v"(voffv), "s"(sbase) : "memory")

__device__ __forceinline__ void wait_async_le8() {
    asm volatile("s_wait_asynccnt 0x8" ::: "memory");
}
__device__ __forceinline__ void wait_async_0() {
    asm volatile("s_wait_asynccnt 0x0" ::: "memory");
}
// flat pointer to __shared__ keeps the LDS byte offset in addr[31:0]
__device__ __forceinline__ unsigned lds_off(const void* p) {
    return (unsigned)(uintptr_t)p;
}

// ---------- kernel 0: pre-convert w_gate (512x7168 fp32) to bf16 hi/lo ----------
__global__ __launch_bounds__(256) void convert_w_kernel(const float* __restrict__ W,
                                                        unsigned short* __restrict__ hiOut,
                                                        unsigned short* __restrict__ loOut) {
    const size_t i = ((size_t)blockIdx.x * 256 + threadIdx.x) * 4;  // 4 floats / thread
    const float4 f = *reinterpret_cast<const float4*>(W + i);
    unsigned h[4], l[4];
    bf16_split(f.x, h[0], l[0]);
    bf16_split(f.y, h[1], l[1]);
    bf16_split(f.z, h[2], l[2]);
    bf16_split(f.w, h[3], l[3]);
    *reinterpret_cast<uint2*>(hiOut + i) = make_uint2(h[0] | (h[1] << 16), h[2] | (h[3] << 16));
    *reinterpret_cast<uint2*>(loOut + i) = make_uint2(l[0] | (l[1] << 16), l[2] | (l[3] << 16));
}

// ---------- kernel 1: kv_score = x @ w_gate^T, bf16x2 split precision ----------
// A (x) converted inline (8 floats/thread/K-step, perm-packed); B (w) async-
// copied from pre-converted bf16 via GLOBAL_LOAD_ASYNC_TO_LDS (saddr form),
// double buffered on ASYNCcnt.
__global__ __launch_bounds__(256) void gemm_bf16x2_kernel(const float* __restrict__ X,
                                                          const unsigned short* __restrict__ Whi,
                                                          const unsigned short* __restrict__ Wlo,
                                                          float* __restrict__ OUT) {
    __shared__ unsigned short Ah[2][BM * LDSS];   //  10 KB
    __shared__ unsigned short Al[2][BM * LDSS];   //  10 KB
    __shared__ unsigned short Bh[2][BN * LDSS];   //  40 KB
    __shared__ unsigned short Bl[2][BN * LDSS];   //  40 KB

    const int t    = threadIdx.x;
    const int lane = t & 31;
    const int wave = t >> 5;
    const int m0 = blockIdx.y * BM;
    const int n0 = blockIdx.x * BN;
    const int mWave = (wave & 1) * 32;   // 2 wave-rows of 32
    const int nWave = (wave >> 1) * 64;  // 4 wave-cols of 64

    // A staging: 4 threads per row, 8 floats each
    const int aRow = t >> 2;
    const int aSeg = t & 3;
    const int aBase = aRow * LDSS + aSeg * 8;
    // B staging: 1 row (32 bf16 hi + 32 bf16 lo) per thread
    const int bRow = t;
    const int bBase = bRow * LDSS;
    // loop-invariant 32-bit byte offset into the bf16 w arrays
    const unsigned bVoff = (unsigned)((size_t)(n0 + bRow) * DIM * 2u);
    // loop-invariant LDS dest offsets (per buffer)
    const unsigned lhB[2] = { lds_off(&Bh[0][bBase]), lds_off(&Bh[1][bBase]) };
    const unsigned llB[2] = { lds_off(&Bl[0][bBase]), lds_off(&Bl[1][bBase]) };

    v8f acc[2][4];
#pragma unroll
    for (int mt = 0; mt < 2; ++mt)
#pragma unroll
        for (int nt = 0; nt < 4; ++nt)
#pragma unroll
            for (int r = 0; r < 8; ++r) acc[mt][nt][r] = 0.0f;

    // --- staging helpers ---
    auto issueB = [&](int kb, int buf) {
        const unsigned short* sbH = Whi + (size_t)kb * BK;   // scalar base (SGPR pair)
        const unsigned short* sbL = Wlo + (size_t)kb * BK;
        const unsigned lh = lhB[buf];
        const unsigned ll = llB[buf];
        ASYNC_B128(lh, bVoff, sbH, "0");
        ASYNC_B128(lh, bVoff, sbH, "16");
        ASYNC_B128(lh, bVoff, sbH, "32");
        ASYNC_B128(lh, bVoff, sbH, "48");
        ASYNC_B128(ll, bVoff, sbL, "0");
        ASYNC_B128(ll, bVoff, sbL, "16");
        ASYNC_B128(ll, bVoff, sbL, "32");
        ASYNC_B128(ll, bVoff, sbL, "48");
    };
    float4 ra[2];
    auto loadA = [&](int kb) {
        const float* ap = X + (size_t)(m0 + aRow) * DIM + kb * BK + aSeg * 8;
        ra[0] = reinterpret_cast<const float4*>(ap)[0];
        ra[1] = reinterpret_cast<const float4*>(ap)[1];
    };
    auto storeA = [&](int buf) {
        const float* f = reinterpret_cast<const float*>(ra);
        unsigned hw[4], lw[4];
#pragma unroll
        for (int i = 0; i < 4; ++i)
            split_pack2(f[2 * i], f[2 * i + 1], hw[i], lw[i]);
        *reinterpret_cast<uint4*>(&Ah[buf][aBase]) = make_uint4(hw[0], hw[1], hw[2], hw[3]);
        *reinterpret_cast<uint4*>(&Al[buf][aBase]) = make_uint4(lw[0], lw[1], lw[2], lw[3]);
    };

    // --- prologue: B(0)->buf0, B(1)->buf1 in flight; A(0) staged, A(1) in regs ---
    issueB(0, 0);
    issueB(1, 1);
    loadA(0);
    storeA(0);
    loadA(1);

    // per-lane fragment addressing (ISA 16-bit A/B layouts, wave32)
    const bool laneLow = (lane < 16);
    const int  rA  = lane & 15;
    const int  aK0 = laneLow ? 0 : 8;    // A: K {0..7,16..23} | {8..15,24..31}
    const int  bK0 = laneLow ? 0 : 8;    // B: K {0..7,16..23} | {8..15,24..31}

    for (int kb = 0; kb < NKB; ++kb) {
        const int buf = kb & 1;

        if (kb + 1 < NKB) wait_async_le8();  // B(kb) landed (in-order completion)
        else              wait_async_0();
        __syncthreads();                     // all waves' B(kb)+A(kb) visible

        // ---- compute on buf ----
        FragU aH[2], aL[2];
#pragma unroll
        for (int mt = 0; mt < 2; ++mt) {
            const int row = mWave + mt * 16 + rA;
            const uint4* ph = reinterpret_cast<const uint4*>(&Ah[buf][row * LDSS + aK0]);
            const uint4* pl = reinterpret_cast<const uint4*>(&Al[buf][row * LDSS + aK0]);
            aH[mt].q[0] = ph[0]; aH[mt].q[1] = ph[2];   // +32B = K+16
            aL[mt].q[0] = pl[0]; aL[mt].q[1] = pl[2];
        }
#pragma unroll
        for (int nt = 0; nt < 4; ++nt) {
            const int row = nWave + nt * 16 + rA;
            FragU bHf, bLf;
            const uint4* ph = reinterpret_cast<const uint4*>(&Bh[buf][row * LDSS + bK0]);
            const uint4* pl = reinterpret_cast<const uint4*>(&Bl[buf][row * LDSS + bK0]);
            bHf.q[0] = ph[0]; bHf.q[1] = ph[2];
            bLf.q[0] = pl[0]; bLf.q[1] = pl[2];
#pragma unroll
            for (int mt = 0; mt < 2; ++mt) {
                acc[mt][nt] = __builtin_amdgcn_wmma_f32_16x16x32_bf16(
                    false, aH[mt].v, false, bHf.v, (short)0, acc[mt][nt], false, false);
                acc[mt][nt] = __builtin_amdgcn_wmma_f32_16x16x32_bf16(
                    false, aL[mt].v, false, bHf.v, (short)0, acc[mt][nt], false, false);
                acc[mt][nt] = __builtin_amdgcn_wmma_f32_16x16x32_bf16(
                    false, aH[mt].v, false, bLf.v, (short)0, acc[mt][nt], false, false);
            }
        }

        __syncthreads();                     // all waves done reading buf
        if (kb + 2 < NKB) issueB(kb + 2, buf);   // safe: buf fully consumed
        if (kb + 1 < NKB) storeA(buf ^ 1);       // stage A(kb+1)
        if (kb + 2 < NKB) loadA(kb + 2);         // prefetch A(kb+2) into regs
    }

    // C layout: VGPR r -> M = r + (lane<16 ? 0 : 8), N = lane%16
#pragma unroll
    for (int mt = 0; mt < 2; ++mt)
#pragma unroll
        for (int nt = 0; nt < 4; ++nt) {
            const int col     = n0 + nWave + nt * 16 + rA;
            const int rowBase = m0 + mWave + mt * 16 + (laneLow ? 0 : 8);
#pragma unroll
            for (int r = 0; r < 8; ++r)
                OUT[(size_t)(rowBase + r) * NOUT + col] = acc[mt][nt][r];
        }
}

// ---------- kernel 2: overlap + softmax-pool + rmsnorm + rope + hadamard ----------
__global__ __launch_bounds__(128) void mqa_epilogue_kernel(const float* __restrict__ KV,
                                                           const float* __restrict__ ape,
                                                           const float* __restrict__ nw,
                                                           const float* __restrict__ cosT,
                                                           const float* __restrict__ sinT,
                                                           float* __restrict__ out) {
    __shared__ float vec[128];
    __shared__ float red[4];
    const int s = blockIdx.x;
    const int d = threadIdx.x;
    const int lane = d & 31;
    const int wv   = d >> 5;

    float kvv[8], sc[8];
#pragma unroll
    for (int j = 0; j < 4; ++j) {
        if (s == 0) { kvv[j] = 0.0f; sc[j] = -1e30f; }
        else {
            const size_t rp = (size_t)((s - 1) * 4 + j) * NOUT;
            kvv[j] = KV[rp + d];          // kv[s-1, j, 0:128]
            sc[j]  = KV[rp + 256 + d];    // score[s-1, j, 0:128]
        }
        const size_t rc = (size_t)(s * 4 + j) * NOUT;
        kvv[4 + j] = KV[rc + 128 + d];    // kv[s, j, 128:256]
        sc[4 + j]  = KV[rc + 384 + d];    // score[s, j, 128:256]
    }
#pragma unroll
    for (int j = 0; j < 8; ++j) kvv[j] += ape[j * 128 + d];

    // softmax over the 8 overlap slots (independent per (s,d)), fused pool
    float m = sc[0];
#pragma unroll
    for (int j = 1; j < 8; ++j) m = fmaxf(m, sc[j]);
    float sum = 0.0f, pooled = 0.0f;
#pragma unroll
    for (int j = 0; j < 8; ++j) {
        const float e = __expf(sc[j] - m);
        sum += e;
        pooled += e * kvv[j];
    }
    pooled /= sum;

    // rmsnorm over 128 dims
    float sq = pooled * pooled;
#pragma unroll
    for (int off = 16; off > 0; off >>= 1) sq += __shfl_xor(sq, off, 32);
    if (lane == 0) red[wv] = sq;
    __syncthreads();
    const float ms = (red[0] + red[1] + red[2] + red[3]) * (1.0f / 128.0f);
    const float normed = pooled * rsqrtf(ms + 1e-6f) * nw[d];
    vec[d] = normed;
    __syncthreads();

    // rope on dims [64,128), pos = s*4+3, half = 32
    const int pos = s * 4 + 3;
    float val;
    if (d < 64) {
        val = normed;
    } else if (d < 96) {
        const float c = cosT[pos * 32 + (d - 64)];
        const float sn = sinT[pos * 32 + (d - 64)];
        val = normed * c - vec[d + 32] * sn;
    } else {
        const float c = cosT[pos * 32 + (d - 96)];
        const float sn = sinT[pos * 32 + (d - 96)];
        val = normed * c + vec[d - 32] * sn;
    }

    // 128-pt Walsh-Hadamard: stages commute; 5 in-wave, 2 via LDS
#pragma unroll
    for (int bs = 1; bs <= 16; bs <<= 1) {
        const float p = __shfl_xor(val, bs, 32);
        val = (d & bs) ? (p - val) : (val + p);
    }
    __syncthreads();
    vec[d] = val;
    __syncthreads();
    { const float p = vec[d ^ 32]; val = (d & 32) ? (p - val) : (val + p); }
    __syncthreads();
    vec[d] = val;
    __syncthreads();
    { const float p = vec[d ^ 64]; val = (d & 64) ? (p - val) : (val + p); }

    out[(size_t)s * 128 + d] = val * 0.08838834764831845f;  // 1/sqrt(128)
}

extern "C" void kernel_launch(void* const* d_in, const int* in_sizes, int n_in,
                              void* d_out, int out_size, void* d_ws, size_t ws_size,
                              hipStream_t stream) {
    (void)in_sizes; (void)n_in; (void)out_size; (void)ws_size;
    const float* x   = (const float*)d_in[0];
    const float* wg  = (const float*)d_in[1];
    const float* ape = (const float*)d_in[2];
    const float* nw  = (const float*)d_in[3];
    const float* ct  = (const float*)d_in[4];
    const float* st  = (const float*)d_in[5];
    float* out = (float*)d_out;

    // workspace layout: kv_score (16 MB) | w_hi bf16 (7.34 MB) | w_lo bf16 (7.34 MB)
    float* kvscore = (float*)d_ws;
    unsigned short* whi = (unsigned short*)((char*)d_ws + (size_t)T_TOK * NOUT * sizeof(float));
    unsigned short* wlo = whi + (size_t)NOUT * DIM;

    const int wElems = NOUT * DIM;                       // 3,670,016
    convert_w_kernel<<<wElems / (4 * 256), 256, 0, stream>>>(wg, whi, wlo);

    dim3 grid(NOUT / BN, T_TOK / BM);                    // (2, 128)
    gemm_bf16x2_kernel<<<grid, 256, 0, stream>>>(x, whi, wlo, kvscore);

    mqa_epilogue_kernel<<<S_OUT, 128, 0, stream>>>(kvscore, ape, nw, ct, st, out);
}
